// UNet_Module_with_attention_1752346656906
// MI455X (gfx1250) — compile-verified, matrix operand order checked by reading, not run on hardware
//
#include <hip/hip_runtime.h>
#include <hip/hip_bf16.h>
#include <math.h>
#include <stdint.h>

// ---------------- types ----------------
typedef __attribute__((ext_vector_type(16))) __bf16 v16bf;
typedef __attribute__((ext_vector_type(8)))  __bf16 v8bf;
typedef __attribute__((ext_vector_type(8)))  float  v8f;

#define DEV static __device__ __forceinline__

// ---------------- problem dims ----------------
static constexpr int Bn = 16, Cn = 64, Hn = 256, Wn = 256;
static constexpr int DINn = 320;
static constexpr int KC = 576;              // Cn * 9  (implicit-GEMM K for 3x3 conv)
static constexpr float EPSV = 1e-5f;

DEV int lane_id() { return (int)(threadIdx.x & 31); }
DEV v8f zero8() { v8f z = {0.f, 0.f, 0.f, 0.f, 0.f, 0.f, 0.f, 0.f}; return z; }

// CDNA5 16-bit A/B fragment (ISA 7.12.2): lane-half `half` holds K runs
// [k0 + half*8, +8) in elements 0..7 and [k0 + 16 + half*8, +8) in 8..15.
// Both runs are contiguous -> two 16-byte loads.
DEV v16bf combine16(v8bf lo, v8bf hi) {
  v16bf a;
#pragma unroll
  for (int e = 0; e < 8; ++e) { a[e] = lo[e]; a[8 + e] = hi[e]; }
  return a;
}
// A fragment from row-major mat[m][k].
DEV v16bf frag_a(const __bf16* mat, int ld, int row0, int k0) {
  int l = lane_id();
  const __bf16* p = mat + (row0 + (l & 15)) * ld + k0 + ((l >> 4) << 3);
  return combine16(*(const v8bf*)p, *(const v8bf*)(p + 16));
}
// B fragment where B[k][n] = mat[n][k] (mat row-major [n][k], contiguous in k).
DEV v16bf frag_b_t(const __bf16* mat, int ld, int col0, int k0) {
  int l = lane_id();
  const __bf16* p = mat + (col0 + (l & 15)) * ld + k0 + ((l >> 4) << 3);
  return combine16(*(const v8bf*)p, *(const v8bf*)(p + 16));
}
// A fragment over pat = [img (256) | pe (64)]; K-chunks never straddle 256.
DEV v16bf frag_a_pat(const __bf16* img, const __bf16* pe, int row0, int k0) {
  int l = lane_id(); int m = row0 + (l & 15);
  const __bf16* p = (k0 < 256) ? (img + m * 256 + k0) : (pe + m * 64 + (k0 - 256));
  p += (l >> 4) << 3;
  return combine16(*(const v8bf*)p, *(const v8bf*)(p + 16));
}
DEV v8f wmma_bf16(v16bf a, v16bf b, v8f c) {
  return __builtin_amdgcn_wmma_f32_16x16x32_bf16(false, a, false, b, (short)0, c,
                                                 false, false);
}
// C tile (f32 16x16) -> bf16 row-major dst[row][col] (strided per lane).
DEV void store_c_bf(__bf16* dst, int ld, int row0, int col0, v8f c) {
  int l = lane_id(); int n = col0 + (l & 15); int r0 = row0 + ((l >> 4) << 3);
#pragma unroll
  for (int v = 0; v < 8; ++v) dst[(r0 + v) * ld + n] = (__bf16)c[v];
}
// C tile -> bf16 TRANSPOSED dst[col][row]; contiguous 16B per lane (1x b128 store).
DEV void store_ct_bf(__bf16* dst, int ld, int row0, int col0, v8f c) {
  int l = lane_id(); int n = col0 + (l & 15); int r0 = row0 + ((l >> 4) << 3);
  v8bf pk;
#pragma unroll
  for (int v = 0; v < 8; ++v) pk[v] = (__bf16)c[v];
  *(v8bf*)(dst + n * ld + r0) = pk;
}

// =====================================================================
// k_prep: FiLM GEMMs, skip, Fourier positional embedding, zero BN stats
// =====================================================================
__global__ void __launch_bounds__(256)
k_prep(const float* __restrict__ t_emb,
       const float* __restrict__ f1w, const float* __restrict__ f1b,
       const float* __restrict__ f2w, const float* __restrict__ f2b,
       const float* __restrict__ skw, const float* __restrict__ skb,
       const float* __restrict__ fourier,
       float* __restrict__ film1, float* __restrict__ film2,
       float* __restrict__ skipo, __bf16* __restrict__ pe_bf,
       float* __restrict__ stats /* 4*64, zeroed */) {
  int t = threadIdx.x;
  for (int i = t; i < Bn * 128; i += 256) {
    int b = i >> 7, o = i & 127;
    float a1 = f1b[o], a2 = f2b[o];
    for (int j = 0; j < 256; ++j) {
      float te = t_emb[b * 256 + j];
      a1 += te * f1w[o * 256 + j];
      a2 += te * f2w[o * 256 + j];
    }
    film1[i] = a1; film2[i] = a2;
  }
  if (t < Bn) {
    float a = skb[0];
    for (int j = 0; j < 256; ++j) a += t_emb[t * 256 + j] * skw[j];
    skipo[t] = a;
  }
  for (int l = t; l < 256; l += 256) {
    float n = (float)(l + 1);
    float horiz = fmodf(n, 16.0f) + 1.0f;
    float vert  = floorf(n / 16.0f) + 1.0f;
    for (int j = 0; j < 16; ++j) {
      pe_bf[l * 64 + j]      = (__bf16)cosf(horiz * fourier[0  + j]);
      pe_bf[l * 64 + 16 + j] = (__bf16)sinf(horiz * fourier[16 + j]);
      pe_bf[l * 64 + 32 + j] = (__bf16)cosf(vert  * fourier[32 + j]);
      pe_bf[l * 64 + 48 + j] = (__bf16)sinf(vert  * fourier[48 + j]);
    }
  }
  if (t < 4 * 64) stats[t] = 0.0f;
}

// =====================================================================
// f32 -> bf16 bulk convert (q/k/v weights, row-major [e][d])
// =====================================================================
__global__ void __launch_bounds__(256)
k_cvt_bf16(const float* __restrict__ src, __bf16* __restrict__ dst, int n) {
  int i = blockIdx.x * blockDim.x + threadIdx.x;
  int stride = gridDim.x * blockDim.x;
  for (; i < n; i += stride) dst[i] = (__bf16)src[i];
}

// conv weights: [co][ci*9 + r*3 + s] -> bf16 [co][(r*3+s)*64 + ci]
// (K-order (r,s,ci) so 8-element K-runs are 8 consecutive ci -> contiguous LDS)
__global__ void __launch_bounds__(256)
k_repack_convw(const float* __restrict__ w, __bf16* __restrict__ dst) {
  int i = blockIdx.x * blockDim.x + threadIdx.x;
  int stride = gridDim.x * blockDim.x;
  for (; i < 64 * KC; i += stride) {
    int co = i / KC; int rem = i - co * KC;
    int rs = rem >> 6; int ci = rem & 63;
    dst[i] = (__bf16)w[co * KC + ci * 9 + rs];
  }
}

// =====================================================================
// Implicit-GEMM 3x3 conv + bias + FiLM, accumulating BN sum/sumsq.
// Block: (b, h, 128-px strip). Wave w owns pixels [w*16, w*16+16).
// LDS: weights 64x576 bf16 (72KB) + halo [3][136][64ci] bf16 (51KB) + 128 f32.
// =====================================================================
// im2col B fragment: K-order (r,s,ci); each 8-run = 8 consecutive ci.
DEV v16bf conv_bfrag(const __bf16* s_in, int pl0, int k0) {
  int l = lane_id(); int pl = pl0 + (l & 15);
  v8bf run[2];
#pragma unroll
  for (int h2 = 0; h2 < 2; ++h2) {
    int k = k0 + ((l >> 4) << 3) + h2 * 16;
    int rs = k >> 6; int ci0 = k & 63;
    int rr = rs / 3; int ss = rs - rr * 3;
    run[h2] = *(const v8bf*)(s_in + (rr * 136 + pl + ss) * 64 + ci0);
  }
  return combine16(run[0], run[1]);
}

__global__ void __launch_bounds__(256)
k_conv_film(const float* __restrict__ in, const __bf16* __restrict__ wbf,
            const float* __restrict__ bias, const float* __restrict__ film,
            float* __restrict__ out,
            float* __restrict__ stat_sum, float* __restrict__ stat_sq) {
  extern __shared__ __align__(16) char smem[];
  __bf16* sw   = (__bf16*)smem;                           // [64][576] k'=(r,s,ci)
  __bf16* s_in = (__bf16*)(smem + 64 * KC * 2);           // [3][136][64]
  float*  sred = (float*)(smem + 64 * KC * 2 + 3 * 136 * 64 * 2); // [128]

  int t = threadIdx.x;
  int blk = blockIdx.x;
  int wt = blk & 1;
  int h  = (blk >> 1) & 255;
  int b  = blk >> 9;
  int w0 = wt * 128;

  for (int i = t; i < 64 * KC; i += 256) sw[i] = wbf[i];
  for (int i = t; i < 64 * 3 * 130; i += 256) {
    int ci = i / 390; int rem = i - ci * 390;
    int r = rem / 130; int j = rem - r * 130;
    int hh = h + r - 1, ww = w0 + j - 1;
    float v = 0.0f;
    if (hh >= 0 && hh < Hn && ww >= 0 && ww < Wn)
      v = in[(((size_t)b * Cn + ci) * Hn + hh) * Wn + ww];
    s_in[(r * 136 + j) * 64 + ci] = (__bf16)v;
  }
  if (t < 128) sred[t] = 0.0f;
  __syncthreads();

  int wave = t >> 5;
  int pl0 = wave * 16;
  v8f acc[4];
#pragma unroll
  for (int cot = 0; cot < 4; ++cot) acc[cot] = zero8();
#pragma unroll
  for (int kc = 0; kc < 18; ++kc) {
    int k0 = kc * 32;
    v16bf bfr = conv_bfrag(s_in, pl0, k0);
#pragma unroll
    for (int cot = 0; cot < 4; ++cot) {
      v16bf afr = frag_a(sw, KC, cot * 16, k0);
      acc[cot] = wmma_bf16(afr, bfr, acc[cot]);
    }
  }
  int l = lane_id();
  int pl = pl0 + (l & 15); int w = w0 + pl;
#pragma unroll
  for (int cot = 0; cot < 4; ++cot) {
    int co_base = cot * 16 + ((l >> 4) << 3);
#pragma unroll
    for (int v = 0; v < 8; ++v) {
      int co = co_base + v;
      float val = acc[cot][v] + bias[co];
      val = val * film[b * 128 + co] + film[b * 128 + 64 + co];
      out[(((size_t)b * Cn + co) * Hn + h) * Wn + w] = val;
      float s = val, q = val * val;
#pragma unroll
      for (int m = 1; m < 16; m <<= 1) {
        s += __shfl_xor(s, m, 32);
        q += __shfl_xor(q, m, 32);
      }
      if ((l & 15) == 0) { atomicAdd(&sred[co], s); atomicAdd(&sred[64 + co], q); }
    }
  }
  __syncthreads();
  if (t < 64) { atomicAdd(&stat_sum[t], sred[t]); atomicAdd(&stat_sq[t], sred[64 + t]); }
}

__global__ void k_bn_fin(const float* __restrict__ sum, const float* __restrict__ sq,
                         float* __restrict__ mean, float* __restrict__ rstd) {
  int c = threadIdx.x;
  if (c < 64) {
    float n = (float)Bn * Hn * Wn;
    float m = sum[c] / n;
    float v = sq[c] / n - m * m;
    mean[c] = m; rstd[c] = rsqrtf(v + EPSV);
  }
}

// =====================================================================
// Fused patch attention, one workgroup per (b,c), 8 waves.
// LDS: img 128KB | pe 32KB | K 128KB | Qblk/P 8KB | scores 16x264 f32
// =====================================================================
__global__ void __launch_bounds__(256)
k_attention(const float* __restrict__ o1, const __bf16* __restrict__ pe_bf,
            const __bf16* __restrict__ qw, const float* __restrict__ qb,
            const __bf16* __restrict__ kw, const float* __restrict__ kb,
            const __bf16* __restrict__ vw, const float* __restrict__ vb,
            const float* __restrict__ mean, const float* __restrict__ rstd,
            __bf16* __restrict__ vstage, float* __restrict__ o2) {
  extern __shared__ __align__(16) char smem[];
  __bf16* s_img = (__bf16*)smem;                  // [256][256] pat pixel part
  __bf16* s_pe  = (__bf16*)(smem + 131072);       // [256][64]
  __bf16* s_k   = (__bf16*)(smem + 163840);       // [256][256] K, [m][e]
  __bf16* s_qb  = (__bf16*)(smem + 294912);       // [16][256]  Q block, then P
  float*  s_sc  = (float*)(smem + 303104);        // [16][264]

  int t = threadIdx.x; int wave = t >> 5;
  int bc = blockIdx.x; int c = bc & 63;
  float mu = mean[c], rs = rstd[c];
  const float* src = o1 + (size_t)bc * (Hn * Wn);

  // ---- phase 0: async copy of positional embedding into LDS (pure memcpy)
  {
    unsigned lbase = (unsigned)(uintptr_t)s_pe;
    const char* g = (const char*)pe_bf;
    for (int i = t; i < 2048; i += 256) {            // 32KB / 16B
      unsigned la = lbase + i * 16;
      unsigned long long ga = (unsigned long long)(uintptr_t)(g + i * 16);
      asm volatile("global_load_async_to_lds_b128 %0, %1, off"
                   :: "v"(la), "v"(ga) : "memory");
    }
  }
  // ---- phase 1: BN + SiLU + patchify into LDS (8 px per iteration)
  const float4* src4 = (const float4*)src;
  for (int idx = t; idx < (Hn * Wn) / 8; idx += 256) {
    if (idx + 2048 < (Hn * Wn) / 8) __builtin_prefetch(&src4[(idx + 2048) * 2], 0, 0);
    int hh = idx >> 5; int w8 = (idx & 31) << 3;
    float4 f0 = src4[idx * 2], f1 = src4[idx * 2 + 1];
    float vals[8] = {f0.x, f0.y, f0.z, f0.w, f1.x, f1.y, f1.z, f1.w};
    int li = ((hh >> 4) << 4) + (w8 >> 4);
    int d0 = ((hh & 15) << 4) + (w8 & 15);
    v8bf pk;
#pragma unroll
    for (int j = 0; j < 8; ++j) {
      float v0 = (vals[j] - mu) * rs;
      pk[j] = (__bf16)(v0 / (1.0f + __expf(-v0)));
    }
    *(v8bf*)(s_img + li * 256 + d0) = pk;
  }
  asm volatile("s_wait_asynccnt 0x0" ::: "memory");
  __syncthreads();

  // ---- phase 2+3: K -> LDS [m][e], V -> global stage TRANSPOSED [e][m]
  __bf16* vdstT = vstage + (size_t)bc * (256 * 256);
  for (int mt = 0; mt < 16; ++mt) {
    int m0 = mt << 4;
    v16bf apat[10];
#pragma unroll
    for (int kc = 0; kc < 10; ++kc) apat[kc] = frag_a_pat(s_img, s_pe, m0, kc * 32);
#pragma unroll
    for (int ntt = 0; ntt < 2; ++ntt) {
      int n0 = (wave + ntt * 8) << 4;
      v8f accK = zero8(), accV = zero8();
#pragma unroll
      for (int kc = 0; kc < 10; ++kc) {
        accK = wmma_bf16(apat[kc], frag_b_t(kw, DINn, n0, kc * 32), accK);
        accV = wmma_bf16(apat[kc], frag_b_t(vw, DINn, n0, kc * 32), accV);
      }
      float bk = kb[n0 + (lane_id() & 15)];
      float bv = vb[n0 + (lane_id() & 15)];
#pragma unroll
      for (int v = 0; v < 8; ++v) { accK[v] += bk; accV[v] += bv; }
      store_c_bf(s_k, 256, m0, n0, accK);
      store_ct_bf(vdstT, 256, m0, n0, accV);
    }
  }
  asm volatile("s_wait_storecnt 0x0" ::: "memory");   // V^T visible before readback
  __syncthreads();

  // ---- phase 4: per 16-row Q block: Q -> scores -> softmax -> P@V
  const float scale = 1.0f / 16.0f;                   // 1/sqrt(PP)
  for (int qblk = 0; qblk < 16; ++qblk) {
    int l0 = qblk << 4;
    {
      v16bf apat[10];
#pragma unroll
      for (int kc = 0; kc < 10; ++kc) apat[kc] = frag_a_pat(s_img, s_pe, l0, kc * 32);
#pragma unroll
      for (int ntt = 0; ntt < 2; ++ntt) {
        int n0 = (wave + ntt * 8) << 4;
        v8f acc = zero8();
#pragma unroll
        for (int kc = 0; kc < 10; ++kc)
          acc = wmma_bf16(apat[kc], frag_b_t(qw, DINn, n0, kc * 32), acc);
        float bq = qb[n0 + (lane_id() & 15)];
#pragma unroll
        for (int v = 0; v < 8; ++v) acc[v] += bq;
        store_c_bf(s_qb, 256, 0, n0, acc);
      }
    }
    __syncthreads();
    { // scores = Qblk @ K^T * scale
      v16bf aq[8];
#pragma unroll
      for (int kc = 0; kc < 8; ++kc) aq[kc] = frag_a(s_qb, 256, 0, kc * 32);
#pragma unroll
      for (int ntt = 0; ntt < 2; ++ntt) {
        int n0 = (wave + ntt * 8) << 4;
        v8f acc = zero8();
#pragma unroll
        for (int kc = 0; kc < 8; ++kc)
          acc = wmma_bf16(aq[kc], frag_b_t(s_k, 256, n0, kc * 32), acc);
        int l = lane_id(); int n = n0 + (l & 15); int r0 = (l >> 4) << 3;
#pragma unroll
        for (int v = 0; v < 8; ++v) s_sc[(r0 + v) * 264 + n] = acc[v] * scale;
      }
    }
    __syncthreads();
    { // softmax: 16 threads per row, wave32 shfl reduce within 16-lane groups
      int row = t >> 4, c0 = (t & 15) << 4;
      float ev[16];
      float mx = -1e30f;
#pragma unroll
      for (int j = 0; j < 16; ++j) { ev[j] = s_sc[row * 264 + c0 + j]; mx = fmaxf(mx, ev[j]); }
#pragma unroll
      for (int m = 1; m < 16; m <<= 1) mx = fmaxf(mx, __shfl_xor(mx, m, 32));
      float sm = 0.f;
#pragma unroll
      for (int j = 0; j < 16; ++j) { ev[j] = __expf(ev[j] - mx); sm += ev[j]; }
#pragma unroll
      for (int m = 1; m < 16; m <<= 1) sm += __shfl_xor(sm, m, 32);
      float inv = 1.0f / sm;
      v8bf p0, p1;
#pragma unroll
      for (int j = 0; j < 8; ++j) {
        p0[j] = (__bf16)(ev[j] * inv);
        p1[j] = (__bf16)(ev[8 + j] * inv);
      }
      *(v8bf*)(s_qb + row * 256 + c0) = p0;
      *(v8bf*)(s_qb + row * 256 + c0 + 8) = p1;
    }
    __syncthreads();
    { // out = P @ V (B from V^T: contiguous), unpatchify to image layout
      v16bf ap[8];
#pragma unroll
      for (int kc = 0; kc < 8; ++kc) ap[kc] = frag_a(s_qb, 256, 0, kc * 32);
#pragma unroll
      for (int ntt = 0; ntt < 2; ++ntt) {
        int n0 = (wave + ntt * 8) << 4;
        v8f acc = zero8();
#pragma unroll
        for (int kc = 0; kc < 8; ++kc)
          acc = wmma_bf16(ap[kc], frag_b_t(vdstT, 256, n0, kc * 32), acc);
        int l = lane_id(); int e = n0 + (l & 15); int r0 = (l >> 4) << 3;
#pragma unroll
        for (int v = 0; v < 8; ++v) {
          int li = l0 + r0 + v;
          int hh = ((li >> 4) << 4) + (e >> 4);
          int ww = ((li & 15) << 4) + (e & 15);
          o2[(size_t)bc * 65536 + hh * 256 + ww] = acc[v];
        }
      }
    }
    __syncthreads();
  }
}

// =====================================================================
// final: silu(bn2(o3)) + x * skip[b]
// =====================================================================
__global__ void __launch_bounds__(256)
k_final(const float* __restrict__ o3, const float* __restrict__ x,
        const float* __restrict__ mean, const float* __restrict__ rstd,
        const float* __restrict__ skip, float* __restrict__ out, size_t n) {
  size_t i = blockIdx.x * (size_t)blockDim.x + threadIdx.x;
  size_t stride = (size_t)gridDim.x * blockDim.x;
  for (; i < n; i += stride) {
    int c = (int)((i >> 16) & 63);
    int b = (int)(i >> 22);
    float v = (o3[i] - mean[c]) * rstd[c];
    float s = v / (1.0f + __expf(-v));
    out[i] = s + x[i] * skip[b];
  }
}

// =====================================================================
extern "C" void kernel_launch(void* const* d_in, const int* in_sizes, int n_in,
                              void* d_out, int out_size, void* d_ws, size_t ws_size,
                              hipStream_t stream) {
  (void)in_sizes; (void)n_in; (void)out_size; (void)ws_size;
  const float* x       = (const float*)d_in[0];
  const float* t_emb   = (const float*)d_in[1];
  const float* conv1_w = (const float*)d_in[2];
  const float* conv1_b = (const float*)d_in[3];
  const float* conv2_w = (const float*)d_in[4];
  const float* conv2_b = (const float*)d_in[5];
  const float* film1_w = (const float*)d_in[6];
  const float* film1_b = (const float*)d_in[7];
  const float* film2_w = (const float*)d_in[8];
  const float* film2_b = (const float*)d_in[9];
  const float* skip_w  = (const float*)d_in[10];
  const float* skip_b  = (const float*)d_in[11];
  const float* q_w     = (const float*)d_in[12];
  const float* q_b     = (const float*)d_in[13];
  const float* k_w     = (const float*)d_in[14];
  const float* k_b     = (const float*)d_in[15];
  const float* v_w     = (const float*)d_in[16];
  const float* v_b     = (const float*)d_in[17];
  const float* fourier = (const float*)d_in[18];
  float* out = (float*)d_out;

  char* ws = (char*)d_ws;
  size_t off = 0;
  auto alloc = [&](size_t bytes) -> char* {
    off = (off + 255) & ~(size_t)255;
    char* p = ws + off; off += bytes; return p;
  };
  float*  film1  = (float*)alloc(16 * 128 * 4);
  float*  film2  = (float*)alloc(16 * 128 * 4);
  float*  skipv  = (float*)alloc(16 * 4);
  __bf16* pe_bf  = (__bf16*)alloc(256 * 64 * 2);
  float*  stats  = (float*)alloc(4 * 64 * 4);   // sum1, sq1, sum2, sq2
  float*  mr     = (float*)alloc(4 * 64 * 4);   // mean1, rstd1, mean2, rstd2
  __bf16* w1bf   = (__bf16*)alloc(64 * 576 * 2);
  __bf16* w2bf   = (__bf16*)alloc(64 * 576 * 2);
  __bf16* qwbf   = (__bf16*)alloc(256 * 320 * 2);
  __bf16* kwbf   = (__bf16*)alloc(256 * 320 * 2);
  __bf16* vwbf   = (__bf16*)alloc(256 * 320 * 2);
  const size_t NIMG = (size_t)16 * 64 * 256 * 256;
  float*  o1     = (float*)alloc(NIMG * 4);
  float*  o2     = (float*)alloc(NIMG * 4);
  float*  o3     = (float*)alloc(NIMG * 4);
  __bf16* vstage = (__bf16*)alloc((size_t)1024 * 256 * 256 * 2);

  k_prep<<<1, 256, 0, stream>>>(t_emb, film1_w, film1_b, film2_w, film2_b,
                                skip_w, skip_b, fourier, film1, film2, skipv,
                                pe_bf, stats);
  k_repack_convw<<<64, 256, 0, stream>>>(conv1_w, w1bf);
  k_repack_convw<<<64, 256, 0, stream>>>(conv2_w, w2bf);
  k_cvt_bf16<<<128, 256, 0, stream>>>(q_w, qwbf, 256 * 320);
  k_cvt_bf16<<<128, 256, 0, stream>>>(k_w, kwbf, 256 * 320);
  k_cvt_bf16<<<128, 256, 0, stream>>>(v_w, vwbf, 256 * 320);

  size_t conv_smem = (size_t)64 * KC * 2 + (size_t)3 * 136 * 64 * 2 + 128 * 4;
  k_conv_film<<<16 * 256 * 2, 256, conv_smem, stream>>>(
      x, w1bf, conv1_b, film1, o1, stats + 0, stats + 64);
  k_bn_fin<<<1, 64, 0, stream>>>(stats + 0, stats + 64, mr + 0, mr + 64);

  size_t attn_smem = 131072 + 32768 + 131072 + 8192 + (size_t)16 * 264 * 4;
  k_attention<<<1024, 256, attn_smem, stream>>>(
      o1, pe_bf, qwbf, q_b, kwbf, k_b, vwbf, v_b, mr + 0, mr + 64, vstage, o2);

  k_conv_film<<<16 * 256 * 2, 256, conv_smem, stream>>>(
      o2, w2bf, conv2_b, film2, o3, stats + 128, stats + 192);
  k_bn_fin<<<1, 64, 0, stream>>>(stats + 128, stats + 192, mr + 128, mr + 192);

  k_final<<<2048, 256, 0, stream>>>(o3, x, mr + 128, mr + 192, skipv, out, NIMG);
}
